// CausalSelfAttention_43078521979239
// MI455X (gfx1250) — matrix operand order repaired and checked by reading, not compiled, not run on hardware
//
#include <hip/hip_runtime.h>

// ---------------------------------------------------------------------------
// Causal self-attention for MI455X (gfx1250, wave32, WMMA bf16 16x16x32).
// B=2, T=2048, C=768, H=12, D=64, 3*H*D=2304.
// GEMMs and attention stage block-shared tiles in LDS via async global->LDS
// copies (ASYNCcnt, gfx1250 builtin), double-buffered.
// ---------------------------------------------------------------------------

#define B_   2
#define T_   2048
#define C_   768
#define H_   12
#define D_   64
#define N3_  2304   // 3*H*D

typedef __attribute__((ext_vector_type(16))) __bf16 v16bf;
typedef __attribute__((ext_vector_type(8)))  __bf16 v8bf;
typedef __attribute__((ext_vector_type(8)))  float  v8f;

// Pointee type must exactly match the async-LDS builtin's parameter:
// 'int __attribute__((vector_size(16)))' in addrspace(1)/(3).
typedef __attribute__((vector_size(16))) int v4i_;
typedef __attribute__((address_space(1))) v4i_* as1_v4i;
typedef __attribute__((address_space(3))) v4i_* as3_v4i;

#if defined(__has_builtin)
#if __has_builtin(__builtin_amdgcn_global_load_async_to_lds_b128)
#define HAVE_ASYNC_LDS 1
#endif
#endif
#ifndef HAVE_ASYNC_LDS
#define HAVE_ASYNC_LDS 0
#endif

static __device__ __forceinline__ __bf16 f2bf(float f) {
  unsigned u = __builtin_bit_cast(unsigned, f);
  unsigned r = (u + 0x7FFFu + ((u >> 16) & 1u)) >> 16;
  unsigned short s = (unsigned short)r;
  return __builtin_bit_cast(__bf16, s);
}

union BF16x16 { v16bf v; v8bf h[2]; };

static __device__ __forceinline__ v16bf load_frag16(const __bf16* p0, const __bf16* p1) {
  BF16x16 f;
  f.h[0] = *(const v8bf*)p0;
  f.h[1] = *(const v8bf*)p1;
  return f.v;
}

#define WMMA_BF16(A, Bm, Cacc) \
  __builtin_amdgcn_wmma_f32_16x16x32_bf16(false, (A), false, (Bm), (short)0, (Cacc), false, false)

// ---- async global -> LDS staging (16B per thread) -------------------------
static __device__ __forceinline__ void stage_b128(const __bf16* g, __bf16* l) {
#if HAVE_ASYNC_LDS
  __builtin_amdgcn_global_load_async_to_lds_b128(
      (as1_v4i)(uintptr_t)g,
      (as3_v4i)(unsigned)(uintptr_t)l,
      0, 0);
#else
  *(v8bf*)l = *(const v8bf*)g;   // synchronous fallback through VGPRs
#endif
}

static __device__ __forceinline__ void wait_async0() {
#if HAVE_ASYNC_LDS
#if __has_builtin(__builtin_amdgcn_s_wait_asynccnt)
  __builtin_amdgcn_s_wait_asynccnt(0);
#else
  asm volatile("s_wait_asynccnt 0x0" ::: "memory");
#endif
#endif
}

static __device__ __forceinline__ void wait_async1() {
#if HAVE_ASYNC_LDS
#if __has_builtin(__builtin_amdgcn_s_wait_asynccnt)
  __builtin_amdgcn_s_wait_asynccnt(1);
#else
  asm volatile("s_wait_asynccnt 0x1" ::: "memory");
#endif
#endif
}

static __device__ __forceinline__ void wait_async2() {
#if HAVE_ASYNC_LDS
#if __has_builtin(__builtin_amdgcn_s_wait_asynccnt)
  __builtin_amdgcn_s_wait_asynccnt(2);
#else
  asm volatile("s_wait_asynccnt 0x2" ::: "memory");
#endif
#endif
}

// ---------------------------------------------------------------------------
// Prep kernels: fp32 -> bf16 convert, and transposed-convert for weights.
// ---------------------------------------------------------------------------
__global__ void k_cvt_x(const float* __restrict__ x, __bf16* __restrict__ xb, int n) {
  int i = blockIdx.x * blockDim.x + threadIdx.x;
  if (i < n) xb[i] = f2bf(x[i]);
}

// w is [K x N] row-major fp32; wt becomes [N x K] row-major bf16.
__global__ void k_transpose(const float* __restrict__ w, __bf16* __restrict__ wt, int K, int N) {
  int i = blockIdx.x * blockDim.x + threadIdx.x;
  if (i < K * N) {
    int nrow = i / K;
    int k    = i % K;
    wt[i] = f2bf(w[(size_t)k * N + nrow]);
  }
}

// ---------------------------------------------------------------------------
// Shared GEMM mainloop: per-wave 16x64 tile of C = A[MxK] * Wt^T, with the
// block-shared 64x32 B tile double-buffered in LDS (async copy) and the A
// fragment software-pipelined one k-step ahead.
// ---------------------------------------------------------------------------
static __device__ __forceinline__ void gemm_tile_mainloop(
    const __bf16* __restrict__ arow,   // A row for this lane (row m0 + lane%16)
    const __bf16* __restrict__ wT,     // [N x K] bf16
    int K, int n0, int tid, __bf16* btile, v8f* acc) {
  const int lane = tid & 31;
  const int half = lane >> 4;
  const int l16  = lane & 15;
  const int srow = tid >> 2;           // 0..63 : B-tile row this thread stages
  const int scol = (tid & 3) * 8;      // 0,8,16,24 : 8-halfs column group
  const __bf16* gsrc = wT + (size_t)(n0 + srow) * K + scol;
  __bf16* ldst = btile + srow * 32 + scol;

  stage_b128(gsrc, ldst);              // prologue: k0 = 0 into buffer 0
  v16bf a = load_frag16(arow + half * 16, arow + 8 + half * 16);

  for (int k0 = 0; k0 < K; k0 += 32) {
    const int cur = (k0 >> 5) & 1;
    __bf16* bcur = btile + cur * (64 * 32);
    __bf16* bnxt = btile + (cur ^ 1) * (64 * 32);
    const bool more = (k0 + 32 < K);
    if (more) {
      stage_b128(gsrc + k0 + 32, bnxt + srow * 32 + scol);
      wait_async1();                   // cur's copy (issued earlier) has landed
    } else {
      wait_async0();
    }
    __syncthreads();                   // all waves' slices of cur are visible

    __builtin_prefetch(arow + k0 + 96, 0, 0);
    v16bf a_next = a;
    if (more)                          // issue next A fragment early (pipelined)
      a_next = load_frag16(arow + k0 + 32 + half * 16, arow + k0 + 40 + half * 16);
#pragma unroll
    for (int nt = 0; nt < 4; ++nt) {
      const __bf16* bl = bcur + (nt * 16 + l16) * 32 + half * 16;
      v16bf b = load_frag16(bl, bl + 8);     // ds_load_b128 x2
      acc[nt] = WMMA_BF16(a, b, acc[nt]);
    }
    __syncthreads();                   // done reading cur before it is refilled
    a = a_next;
  }
}

// ---------------------------------------------------------------------------
// QKV GEMM: qkv[m,n] = sum_k xb[m,k] * Wqkv[k,n], via wT = Wqkv^T [N3 x C].
// Scatter into Q [B,H,T,D], K [B,H,T,D], V^T [B,H,D,T] (all bf16).
// grid (N3/64=36, M/128=32), 256 threads = 8 waves, wave does 16x64 tile.
// ---------------------------------------------------------------------------
__global__ void __launch_bounds__(256)
k_qkv_gemm(const __bf16* __restrict__ xb, const __bf16* __restrict__ wT,
           __bf16* __restrict__ Qb, __bf16* __restrict__ Kb, __bf16* __restrict__ Vt) {
  __shared__ __bf16 btile[2][64 * 32];
  const int tid  = threadIdx.x;
  const int lane = tid & 31;
  const int wave = tid >> 5;
  const int half = lane >> 4;
  const int l16  = lane & 15;
  const int m0 = blockIdx.y * 128 + wave * 16;
  const int n0 = blockIdx.x * 64;

  v8f acc[4] = {};
  const __bf16* arow = xb + (size_t)(m0 + l16) * C_;
  gemm_tile_mainloop(arow, wT, C_, n0, tid, &btile[0][0], acc);

#pragma unroll
  for (int nt = 0; nt < 4; ++nt) {
#pragma unroll
    for (int r = 0; r < 8; ++r) {
      int m = m0 + r + half * 8;
      int n = n0 + nt * 16 + l16;
      int bidx = m / T_, t = m % T_;
      int s = n / C_, c = n % C_;
      int hh = c / D_, d = c % D_;
      __bf16 val = f2bf(acc[nt][r]);
      size_t bh = (size_t)(bidx * H_ + hh);
      if (s == 0)      Qb[(bh * T_ + t) * D_ + d] = val;
      else if (s == 1) Kb[(bh * T_ + t) * D_ + d] = val;
      else             Vt[(bh * D_ + d) * T_ + t] = val;
    }
  }
}

// ---------------------------------------------------------------------------
// Flash-style causal attention with block-shared K/V tiles in LDS.
// grid (T/128=16, B*H=24), 256 threads = 8 waves; wave w owns q rows
// [base+16w, base+16w+16). The kv loop is BLOCK-uniform (to base+128) so all
// waves can cooperatively stage K (32x64) and V^T (64x32) tiles, async +
// double-buffered; waves past their causal limit just pass the barriers
// (they'd be waiting on wave 7 anyway). Online softmax in f32; P restaged
// C-layout -> A-layout through per-wave LDS.
// ---------------------------------------------------------------------------
__global__ void __launch_bounds__(256)
k_attn(const __bf16* __restrict__ Qb, const __bf16* __restrict__ Kb,
       const __bf16* __restrict__ Vt, __bf16* __restrict__ AO) {
  __shared__ __bf16 kbuf[2][32 * 64];   // 2 x 4KB : K rows jb..jb+31
  __shared__ __bf16 vbuf[2][64 * 32];   // 2 x 4KB : V^T cols jb..jb+31
  __shared__ __bf16 pbuf[8][16 * 32];   // 8KB    : per-wave P restage

  const int tid  = threadIdx.x;
  const int lane = tid & 31;
  const int wave = tid >> 5;
  const int half = lane >> 4;
  const int l16  = lane & 15;
  const int bh   = blockIdx.y;
  const int bidx = bh / H_, h = bh % H_;
  const int base  = blockIdx.x * 128;
  const int qbase = base + wave * 16;

  const __bf16* Qh = Qb + (size_t)bh * T_ * D_;
  const __bf16* Kh = Kb + (size_t)bh * T_ * D_;
  const __bf16* Vh = Vt + (size_t)bh * D_ * T_;

  // cooperative staging coordinates
  const int krow = tid >> 3;            // 0..31
  const int kcol = (tid & 7) * 8;       // 0..56
  const int vrow = tid >> 2;            // 0..63
  const int vcol = (tid & 3) * 8;       // 0..24

  // Q A-fragments for the whole tile (row = qbase + l16), D=64 -> two 32-k chunks
  const __bf16* qrow = Qh + (size_t)(qbase + l16) * D_;
  v16bf aQ0 = load_frag16(qrow +  0 + half * 16, qrow +  8 + half * 16);
  v16bf aQ1 = load_frag16(qrow + 32 + half * 16, qrow + 40 + half * 16);

  float mi[8], li[8];
#pragma unroll
  for (int r = 0; r < 8; ++r) { mi[r] = -3.0e38f; li[r] = 0.0f; }
  v8f o0 = {}, o1 = {}, o2 = {}, o3 = {};

  const int wave_jmax = qbase + 16;     // causal limit for this wave
  const int blk_jmax  = base + 128;     // block-uniform loop bound

  // prologue: stage jb=0 tiles into buffer 0 (2 async issues per wave)
  stage_b128(Kh + (size_t)krow * D_ + kcol, &kbuf[0][krow * 64 + kcol]);
  stage_b128(Vh + (size_t)vrow * T_ + vcol, &vbuf[0][vrow * 32 + vcol]);

  for (int jb = 0; jb < blk_jmax; jb += 32) {
    const int cur = (jb >> 5) & 1;
    if (jb + 32 < blk_jmax) {
      stage_b128(Kh + (size_t)(jb + 32 + krow) * D_ + kcol, &kbuf[cur ^ 1][krow * 64 + kcol]);
      stage_b128(Vh + (size_t)vrow * T_ + jb + 32 + vcol,   &vbuf[cur ^ 1][vrow * 32 + vcol]);
      wait_async2();                    // the 2 copies filling cur have landed
    } else {
      wait_async0();
    }
    __syncthreads();

    if (jb < wave_jmax) {               // wave-uniform: EXEC stays all-ones
      v8f s0 = {}, s1 = {};
      {
        const __bf16* kb0 = &kbuf[cur][l16 * 64 + half * 16];
        s0 = WMMA_BF16(aQ0, load_frag16(kb0, kb0 + 8), s0);
        s0 = WMMA_BF16(aQ1, load_frag16(kb0 + 32, kb0 + 40), s0);
        const __bf16* kb1 = &kbuf[cur][(16 + l16) * 64 + half * 16];
        s1 = WMMA_BF16(aQ0, load_frag16(kb1, kb1 + 8), s1);
        s1 = WMMA_BF16(aQ1, load_frag16(kb1 + 32, kb1 + 40), s1);
      }

      // scale, causal mask, online softmax (row = r + half*8 across lanes 0..15)
#pragma unroll
      for (int r = 0; r < 8; ++r) {
        int M = r + half * 8;
        int q = qbase + M;
        float v0 = s0[r] * 0.125f;      // 1/sqrt(64)
        float v1 = s1[r] * 0.125f;
        if (jb + l16 > q)      v0 = -3.0e38f;
        if (jb + 16 + l16 > q) v1 = -3.0e38f;
        float mx = fmaxf(v0, v1);
#pragma unroll
        for (int d = 1; d < 16; d <<= 1) mx = fmaxf(mx, __shfl_xor(mx, d, 32));
        float mnew = fmaxf(mi[r], mx);
        float p0 = __expf(v0 - mnew);
        float p1 = __expf(v1 - mnew);
        float rs = p0 + p1;
#pragma unroll
        for (int d = 1; d < 16; d <<= 1) rs += __shfl_xor(rs, d, 32);
        float sc = __expf(mi[r] - mnew);
        li[r] = li[r] * sc + rs;
        mi[r] = mnew;
        o0[r] *= sc; o1[r] *= sc; o2[r] *= sc; o3[r] *= sc;
        pbuf[wave][M * 32 + l16]      = f2bf(p0);
        pbuf[wave][M * 32 + 16 + l16] = f2bf(p1);
      }

      // cross-lane LDS store -> load within the wave: drain DS counter
      asm volatile("s_wait_dscnt 0" ::: "memory");

      // P as A-fragment (16x32): row = l16, K = (half*16 + e)
      const __bf16* pr = &pbuf[wave][l16 * 32 + half * 16];
      v16bf aP = load_frag16(pr, pr + 8);

      // O += P * V  (V^T rows in LDS are output columns)
      {
        const __bf16* vb0 = &vbuf[cur][( 0 + l16) * 32 + half * 16];
        o0 = WMMA_BF16(aP, load_frag16(vb0, vb0 + 8), o0);
        const __bf16* vb1 = &vbuf[cur][(16 + l16) * 32 + half * 16];
        o1 = WMMA_BF16(aP, load_frag16(vb1, vb1 + 8), o1);
        const __bf16* vb2 = &vbuf[cur][(32 + l16) * 32 + half * 16];
        o2 = WMMA_BF16(aP, load_frag16(vb2, vb2 + 8), o2);
        const __bf16* vb3 = &vbuf[cur][(48 + l16) * 32 + half * 16];
        o3 = WMMA_BF16(aP, load_frag16(vb3, vb3 + 8), o3);
      }
    }
    __syncthreads();                    // block done with cur before refill
  }

  // normalize and write attention output [B*T, H*D] bf16 (A matrix for proj GEMM)
  const size_t aobase = (size_t)(bidx * T_ + qbase) * (H_ * D_) + h * D_;
#pragma unroll
  for (int r = 0; r < 8; ++r) {
    float invr = 1.0f / li[r];
    size_t rowoff = aobase + (size_t)(r + half * 8) * (H_ * D_) + l16;
    AO[rowoff +  0] = f2bf(o0[r] * invr);
    AO[rowoff + 16] = f2bf(o1[r] * invr);
    AO[rowoff + 32] = f2bf(o2[r] * invr);
    AO[rowoff + 48] = f2bf(o3[r] * invr);
  }
}

// ---------------------------------------------------------------------------
// Output projection: out[m,n] = sum_k AO[m,k] * Wproj[k,n], wT = Wproj^T.
// grid (C/64=12, M/128=32), f32 output.
// ---------------------------------------------------------------------------
__global__ void __launch_bounds__(256)
k_proj_gemm(const __bf16* __restrict__ A, const __bf16* __restrict__ wT,
            float* __restrict__ out) {
  __shared__ __bf16 btile[2][64 * 32];
  const int tid  = threadIdx.x;
  const int lane = tid & 31;
  const int wave = tid >> 5;
  const int half = lane >> 4;
  const int l16  = lane & 15;
  const int m0 = blockIdx.y * 128 + wave * 16;
  const int n0 = blockIdx.x * 64;

  v8f acc[4] = {};
  const __bf16* arow = A + (size_t)(m0 + l16) * (H_ * D_);
  gemm_tile_mainloop(arow, wT, H_ * D_, n0, tid, &btile[0][0], acc);

#pragma unroll
  for (int nt = 0; nt < 4; ++nt) {
#pragma unroll
    for (int r = 0; r < 8; ++r) {
      int m = m0 + r + half * 8;
      int n = n0 + nt * 16 + l16;
      out[(size_t)m * C_ + n] = acc[nt][r];
    }
  }
}

// ---------------------------------------------------------------------------
// Host launcher
// ---------------------------------------------------------------------------
extern "C" void kernel_launch(void* const* d_in, const int* in_sizes, int n_in,
                              void* d_out, int out_size, void* d_ws, size_t ws_size,
                              hipStream_t stream) {
  (void)in_sizes; (void)n_in; (void)out_size; (void)ws_size;
  const float* x     = (const float*)d_in[0];   // [B,T,C]
  const float* Wqkv  = (const float*)d_in[1];   // [C, 3HD]
  const float* Wproj = (const float*)d_in[2];   // [HD, C]
  float* out = (float*)d_out;                   // [B,T,C]

  char* ws = (char*)d_ws;
  size_t off = 0;
  __bf16* xb     = (__bf16*)(ws + off); off += (size_t)B_ * T_ * C_ * 2;       // 6.29 MB
  __bf16* wqkvT  = (__bf16*)(ws + off); off += (size_t)N3_ * C_ * 2;           // 3.54 MB
  __bf16* wprojT = (__bf16*)(ws + off); off += (size_t)C_ * C_ * 2;            // 1.18 MB
  __bf16* Qb     = (__bf16*)(ws + off); off += (size_t)B_ * H_ * T_ * D_ * 2;  // 6.29 MB
  __bf16* Kb     = (__bf16*)(ws + off); off += (size_t)B_ * H_ * T_ * D_ * 2;  // 6.29 MB
  __bf16* Vt     = (__bf16*)(ws + off); off += (size_t)B_ * H_ * D_ * T_ * 2;  // 6.29 MB
  __bf16* AO     = (__bf16*)(ws + off); off += (size_t)B_ * T_ * (H_ * D_) * 2;// 6.29 MB

  const int nX = B_ * T_ * C_;
  k_cvt_x<<<(nX + 255) / 256, 256, 0, stream>>>(x, xb, nX);
  k_transpose<<<(C_ * N3_ + 255) / 256, 256, 0, stream>>>(Wqkv, wqkvT, C_, N3_);
  k_transpose<<<(C_ * C_ + 255) / 256, 256, 0, stream>>>(Wproj, wprojT, C_, C_);

  k_qkv_gemm<<<dim3(N3_ / 64, (B_ * T_) / 128), 256, 0, stream>>>(xb, wqkvT, Qb, Kb, Vt);
  k_attn<<<dim3(T_ / 128, B_ * H_), 256, 0, stream>>>(Qb, Kb, Vt, AO);
  k_proj_gemm<<<dim3(C_ / 64, (B_ * T_) / 128), 256, 0, stream>>>(AO, wprojT, out);
}
